// GIN_42975442764308
// MI455X (gfx1250) — matrix operand order repaired and checked by reading, not compiled, not run on hardware
//
#include <hip/hip_runtime.h>

typedef __attribute__((ext_vector_type(16))) _Float16 v16h;
typedef __attribute__((ext_vector_type(8)))  float    v8f;

#define N_NODES 10000
#define N_EDGES 80000
#define AB 2
#define CT 15
#define DF 64
#define ROWS (N_NODES * AB * CT)   // 300000 matmul rows
#define OUTD 10
#define NGIN 2

// ---------------------------------------------------------------------------
// h [A,D,C,N] -> X [N,A,C,D]
__global__ __launch_bounds__(256) void transpose_h(const float* __restrict__ h,
                                                   float* __restrict__ X, long total) {
    long stride = (long)blockDim.x * gridDim.x;
    for (long idx = (long)blockIdx.x * blockDim.x + threadIdx.x; idx < total; idx += stride) {
        int  d  = (int)(idx & 63);
        long r  = idx >> 6;          // ((n*A + a)*C + c)
        int  c  = (int)(r % CT);
        long r2 = r / CT;
        int  a  = (int)(r2 & 1);
        long n  = r2 >> 1;
        X[idx] = h[(((long)a * DF + d) * CT + c) * N_NODES + n];
    }
}

// ---------------------------------------------------------------------------
// agg[dst] += x[src] over the 1920-float node slab
__global__ __launch_bounds__(256) void edge_agg(const float* __restrict__ X,
                                                float* __restrict__ AGG,
                                                const int* __restrict__ src,
                                                const int* __restrict__ dst) {
    int e = blockIdx.x;
    int s = src[e], t = dst[e];
    const float* xs = X   + (long)s * (AB * CT * DF);
    float*       ad = AGG + (long)t * (AB * CT * DF);
    for (int i = threadIdx.x; i < AB * CT * DF; i += blockDim.x)
        atomicAdd(&ad[i], xs[i]);
}

// ---------------------------------------------------------------------------
// GEMM1: Z = (1+eps)*X + AGG (fused on A load); ZIO <- Z @ W + bias (in place,
// ZIO holds AGG on entry). Fused per-channel sum / sumsq for BN stats.
__global__ __launch_bounds__(256) void gemm_gin1(const float* __restrict__ X,
                                                 float* ZIO,            // agg in, t out (aliased, no restrict)
                                                 const float* __restrict__ W,
                                                 const float* __restrict__ bias,
                                                 const float* __restrict__ epsp,
                                                 float* __restrict__ gsum,
                                                 float* __restrict__ gsq) {
    __shared__ float s_sum[64], s_sq[64];
    int tid = threadIdx.x;
    if (tid < 64) { s_sum[tid] = 0.f; s_sq[tid] = 0.f; }
    __syncthreads();

    int  wave = tid >> 5, lane = tid & 31;
    int  m  = lane & 15, hi = lane >> 4;
    long rowBase = ((long)blockIdx.x * 8 + wave) * 16;

    if (rowBase < ROWS) {                                  // wave-uniform guard
        float epsv = 1.0f + epsp[0];
        const float* xr = X   + (rowBase + m) * DF;
        float*       ar = ZIO + (rowBase + m) * DF;
        int kb = hi * 8;

        v8f acc[4] = {v8f{}, v8f{}, v8f{}, v8f{}};
        #pragma unroll
        for (int kk = 0; kk < 2; ++kk) {
            int base = kb + kk * 32;
            v16h av;
            #pragma unroll
            for (int i = 0; i < 8; ++i) {
                av[i]     = (_Float16)fmaf(xr[base + i],      epsv, ar[base + i]);
                av[8 + i] = (_Float16)fmaf(xr[base + 16 + i], epsv, ar[base + 16 + i]);
            }
            int kB = kk * 32 + hi * 8;
            #pragma unroll
            for (int g = 0; g < 4; ++g) {
                int n = g * 16 + m;
                v16h bv;
                #pragma unroll
                for (int i = 0; i < 16; ++i)
                    bv[i] = (_Float16)W[(kB + i + (i >= 8 ? 8 : 0)) * DF + n];
                acc[g] = __builtin_amdgcn_wmma_f32_16x16x32_f16(
                    false, av, false, bv, (short)0, acc[g], false, false);
            }
        }
        #pragma unroll
        for (int g = 0; g < 4; ++g) {
            int n = g * 16 + m;
            float bb = bias[n];
            float ps = 0.f, pq = 0.f;
            #pragma unroll
            for (int v = 0; v < 8; ++v) {
                float val = acc[g][v] + bb;
                ZIO[(rowBase + hi * 8 + v) * DF + n] = val;
                ps += val; pq += val * val;
            }
            atomicAdd(&s_sum[n], ps);
            atomicAdd(&s_sq[n],  pq);
        }
    }
    __syncthreads();
    if (tid < 64) { atomicAdd(&gsum[tid], s_sum[tid]); atomicAdd(&gsq[tid], s_sq[tid]); }
}

// ---------------------------------------------------------------------------
// GEMM2: A = relu(T*sc + sh) fused on load; ZIO <- A @ W + bias (in place).
// Fused per-channel stats of the output.
__global__ __launch_bounds__(256) void gemm_gin2(float* ZIO,            // t in, u out (aliased)
                                                 const float* __restrict__ sc,
                                                 const float* __restrict__ sh,
                                                 const float* __restrict__ W,
                                                 const float* __restrict__ bias,
                                                 float* __restrict__ gsum,
                                                 float* __restrict__ gsq) {
    __shared__ float s_sum[64], s_sq[64];
    int tid = threadIdx.x;
    if (tid < 64) { s_sum[tid] = 0.f; s_sq[tid] = 0.f; }
    __syncthreads();

    int  wave = tid >> 5, lane = tid & 31;
    int  m  = lane & 15, hi = lane >> 4;
    long rowBase = ((long)blockIdx.x * 8 + wave) * 16;

    if (rowBase < ROWS) {
        float* tr = ZIO + (rowBase + m) * DF;
        int kb = hi * 8;

        v8f acc[4] = {v8f{}, v8f{}, v8f{}, v8f{}};
        #pragma unroll
        for (int kk = 0; kk < 2; ++kk) {
            int base = kb + kk * 32;
            v16h av;
            #pragma unroll
            for (int i = 0; i < 8; ++i) {
                int k0 = base + i, k1 = base + 16 + i;
                av[i]     = (_Float16)fmaxf(fmaf(tr[k0], sc[k0], sh[k0]), 0.f);
                av[8 + i] = (_Float16)fmaxf(fmaf(tr[k1], sc[k1], sh[k1]), 0.f);
            }
            int kB = kk * 32 + hi * 8;
            #pragma unroll
            for (int g = 0; g < 4; ++g) {
                int n = g * 16 + m;
                v16h bv;
                #pragma unroll
                for (int i = 0; i < 16; ++i)
                    bv[i] = (_Float16)W[(kB + i + (i >= 8 ? 8 : 0)) * DF + n];
                acc[g] = __builtin_amdgcn_wmma_f32_16x16x32_f16(
                    false, av, false, bv, (short)0, acc[g], false, false);
            }
        }
        #pragma unroll
        for (int g = 0; g < 4; ++g) {
            int n = g * 16 + m;
            float bb = bias[n];
            float ps = 0.f, pq = 0.f;
            #pragma unroll
            for (int v = 0; v < 8; ++v) {
                float val = acc[g][v] + bb;
                ZIO[(rowBase + hi * 8 + v) * DF + n] = val;
                ps += val; pq += val * val;
            }
            atomicAdd(&s_sum[n], ps);
            atomicAdd(&s_sq[n],  pq);
        }
    }
    __syncthreads();
    if (tid < 64) { atomicAdd(&gsum[tid], s_sum[tid]); atomicAdd(&gsq[tid], s_sq[tid]); }
}

// ---------------------------------------------------------------------------
// per-channel BN coefficients from batch sums: y = x*sc + sh
__global__ void bn_coef(const float* __restrict__ gsum, const float* __restrict__ gsq,
                        const float* __restrict__ gamma, const float* __restrict__ beta,
                        float invM, float* __restrict__ sc, float* __restrict__ sh) {
    int n = threadIdx.x;                       // 64 threads
    float mean = gsum[n] * invM;
    float var  = gsq[n] * invM - mean * mean;  // biased var, as torch forward
    float rs   = rsqrtf(var + 1e-5f);
    float s    = gamma[n] * rs;
    sc[n] = s;
    sh[n] = beta[n] - mean * s;
}

// ---------------------------------------------------------------------------
// U <- relu(U*sc + sh) in place, with fused stats of the result
__global__ __launch_bounds__(256) void ew_bnrelu_stats(float* __restrict__ U,
                                                       const float* __restrict__ sc,
                                                       const float* __restrict__ sh,
                                                       float* __restrict__ gsum,
                                                       float* __restrict__ gsq, long total) {
    __shared__ float s_sum[64], s_sq[64];
    int tid = threadIdx.x;
    if (tid < 64) { s_sum[tid] = 0.f; s_sq[tid] = 0.f; }
    __syncthreads();
    int   ch = tid & 63;                 // stride is a multiple of 64 -> channel fixed
    float s = sc[ch], b = sh[ch];
    float ps = 0.f, pq = 0.f;
    long stride = (long)blockDim.x * gridDim.x;
    for (long i = (long)blockIdx.x * blockDim.x + tid; i < total; i += stride) {
        float z = fmaxf(fmaf(U[i], s, b), 0.f);
        U[i] = z;
        ps += z; pq += z * z;
    }
    atomicAdd(&s_sum[ch], ps);
    atomicAdd(&s_sq[ch],  pq);
    __syncthreads();
    if (tid < 64) { atomicAdd(&gsum[tid], s_sum[tid]); atomicAdd(&gsq[tid], s_sq[tid]); }
}

// X <- relu(U*sc + sh)
__global__ __launch_bounds__(256) void ew_bnrelu(const float* __restrict__ U,
                                                 const float* __restrict__ sc,
                                                 const float* __restrict__ sh,
                                                 float* __restrict__ X, long total) {
    int   ch = threadIdx.x & 63;
    float s = sc[ch], b = sh[ch];
    long stride = (long)blockDim.x * gridDim.x;
    for (long i = (long)blockIdx.x * blockDim.x + threadIdx.x; i < total; i += stride)
        X[i] = fmaxf(fmaf(U[i], s, b), 0.f);
}

// ---------------------------------------------------------------------------
// pooled[a,c,d] = sum_n rep[n,a,c,d]
__global__ void pool_nodes(const float* __restrict__ rep, float* __restrict__ pooled) {
    int ac = blockIdx.x;                 // 0..29  (a*C + c)
    int sl = blockIdx.y;                 // 0..63  node slice
    int d  = threadIdx.x;                // 0..63
    int n0 = sl * 157;
    int n1 = n0 + 157; if (n1 > N_NODES) n1 = N_NODES;
    float acc = 0.f;
    for (int n = n0; n < n1; ++n)
        acc += rep[((long)n * (AB * CT) + ac) * DF + d];
    atomicAdd(&pooled[ac * DF + d], acc);
}

// ---------------------------------------------------------------------------
// score = sum_i pooled_i @ Wp[i] + bp[i];  out[a] = sum_{c,h} score*Wfc + bfc
__global__ void final_readout(const float* __restrict__ pooled,
                              const float* __restrict__ Wp, const float* __restrict__ bp,
                              const float* __restrict__ Wfc, const float* __restrict__ bfc,
                              float* __restrict__ out) {
    __shared__ float s_out[AB];
    int tid = threadIdx.x;
    if (tid < AB) s_out[tid] = 0.f;
    __syncthreads();
    if (tid < AB * CT * OUTD) {
        int a = tid / (CT * OUTD);
        int r = tid % (CT * OUTD);
        int c = r / OUTD, o = r % OUTD;
        float score = 0.f;
        for (int i = 0; i < NGIN + 1; ++i) {
            float accv = bp[i * OUTD + o];
            const float* p = pooled + i * (AB * CT * DF) + (a * CT + c) * DF;
            const float* w = Wp + i * (DF * OUTD);
            for (int d = 0; d < DF; ++d) accv += p[d] * w[d * OUTD + o];
            score += accv;
        }
        atomicAdd(&s_out[a], score * Wfc[c * OUTD + o]);
    }
    __syncthreads();
    if (tid < AB) out[tid] = s_out[tid] + bfc[0];
}

// ---------------------------------------------------------------------------
extern "C" void kernel_launch(void* const* d_in, const int* in_sizes, int n_in,
                              void* d_out, int out_size, void* d_ws, size_t ws_size,
                              hipStream_t stream) {
    const float* h   = (const float*)d_in[0];
    const float* eps = (const float*)d_in[1];
    const float* W0  = (const float*)d_in[2];
    const float* b0  = (const float*)d_in[3];
    const float* g0  = (const float*)d_in[4];
    const float* be0 = (const float*)d_in[5];
    const float* W1  = (const float*)d_in[6];
    const float* b1  = (const float*)d_in[7];
    const float* ga  = (const float*)d_in[8];
    const float* ba  = (const float*)d_in[9];
    const float* go  = (const float*)d_in[10];
    const float* bo  = (const float*)d_in[11];
    const float* Wp  = (const float*)d_in[12];
    const float* bp  = (const float*)d_in[13];
    const float* Wfc = (const float*)d_in[14];
    const float* bfc = (const float*)d_in[15];
    const int*   src = (const int*)d_in[16];
    const int*   dst = (const int*)d_in[17];
    float* out = (float*)d_out;

    const long   TOT   = (long)ROWS * DF;          // 19.2M floats
    float* X      = (float*)d_ws;                  // 76.8 MB
    float* WORK   = X + TOT;                       // 76.8 MB (agg -> t -> u -> z, in place)
    float* stats  = WORK + TOT;                    // 6*64
    float* coef   = stats + 6 * 64;                // 6*64
    float* pooled = coef + 6 * 64;                 // 3*1920

    hipMemsetAsync(pooled, 0, 3 * AB * CT * DF * sizeof(float), stream);

    transpose_h<<<4096, 256, 0, stream>>>(h, X, TOT);
    dim3 pg(AB * CT, 64);
    pool_nodes<<<pg, 64, 0, stream>>>(X, pooled);

    const float invM = 1.0f / (float)ROWS;
    const int gemmGrid = (ROWS / 16 + 7) / 8;      // 8 waves/block, 16 rows/wave

    for (int l = 0; l < NGIN; ++l) {
        hipMemsetAsync(WORK,  0, TOT * sizeof(float), stream);     // agg = 0
        hipMemsetAsync(stats, 0, 6 * 64 * sizeof(float), stream);

        edge_agg<<<N_EDGES, 256, 0, stream>>>(X, WORK, src, dst);

        gemm_gin1<<<gemmGrid, 256, 0, stream>>>(X, WORK, W0 + l * DF * DF, b0 + l * DF,
                                                eps + l, stats, stats + 64);
        bn_coef<<<1, 64, 0, stream>>>(stats, stats + 64, g0 + l * DF, be0 + l * DF,
                                      invM, coef, coef + 64);

        gemm_gin2<<<gemmGrid, 256, 0, stream>>>(WORK, coef, coef + 64,
                                                W1 + l * DF * DF, b1 + l * DF,
                                                stats + 128, stats + 192);
        bn_coef<<<1, 64, 0, stream>>>(stats + 128, stats + 192, ga + l * DF, ba + l * DF,
                                      invM, coef + 128, coef + 192);

        ew_bnrelu_stats<<<1024, 256, 0, stream>>>(WORK, coef + 128, coef + 192,
                                                  stats + 256, stats + 320, TOT);
        bn_coef<<<1, 64, 0, stream>>>(stats + 256, stats + 320, go + l * DF, bo + l * DF,
                                      invM, coef + 256, coef + 320);

        ew_bnrelu<<<4096, 256, 0, stream>>>(WORK, coef + 256, coef + 320, X, TOT);

        pool_nodes<<<pg, 64, 0, stream>>>(X, pooled + (l + 1) * (AB * CT * DF));
    }

    final_readout<<<1, 512, 0, stream>>>(pooled, Wp, bp, Wfc, bfc, out);
}
